// Time_Attention_23192823398859
// MI455X (gfx1250) — matrix operand order repaired and checked by reading, not compiled, not run on hardware
//
#include <hip/hip_runtime.h>
#include <hip/hip_bf16.h>

typedef _Float16 h8   __attribute__((ext_vector_type(8)));
typedef _Float16 v16h __attribute__((ext_vector_type(16)));
typedef float    v8f  __attribute__((ext_vector_type(8)));
typedef float    f4   __attribute__((ext_vector_type(4)));

#define N_TOK   1024
#define D_IN    192
#define D_MODEL 512
#define NBATCH  64          // B*D = 8*8
#define INV_T   20.0f       // 1 / 0.05
#define LNEG    (-23.025850929940457f)   // -log(1e10)
#define SCALE   0.044194173824159216f    // 1/sqrt(512)

union frag16 { v16h v; h8 h[2]; };

static __device__ __forceinline__ v8f wmma_f16(v16h a, v16h b, v8f c) {
    return __builtin_amdgcn_wmma_f32_16x16x32_f16(
        /*neg_a=*/false, a, /*neg_b=*/false, b,
        /*c_mod=*/(short)0, c, /*reuse_a=*/false, /*reuse_b=*/false);
}

static __device__ __forceinline__ h8 cvt_h8(f4 a, f4 b) {
    h8 h;
    h[0] = (_Float16)a[0]; h[1] = (_Float16)a[1];
    h[2] = (_Float16)a[2]; h[3] = (_Float16)a[3];
    h[4] = (_Float16)b[0]; h[5] = (_Float16)b[1];
    h[6] = (_Float16)b[2]; h[7] = (_Float16)b[3];
    return h;
}

// Load an A-fragment (16x32 f16) for lane `lane` from a row-major f32 matrix.
// Per ISA layout: lane holds row (lane&15); half-chunks {sel*8..+7, 16+sel*8..+7}.
static __device__ __forceinline__ frag16 load_afrag_f32(const float* rowp, int sel) {
    f4 c0 = *(const f4*)(rowp + sel * 8);
    f4 c1 = *(const f4*)(rowp + sel * 8 + 4);
    f4 c2 = *(const f4*)(rowp + 16 + sel * 8);
    f4 c3 = *(const f4*)(rowp + 16 + sel * 8 + 4);
    frag16 f;
    f.h[0] = cvt_h8(c0, c1);
    f.h[1] = cvt_h8(c2, c3);
    return f;
}

// Load a B-fragment (32x16 f16): lane holds column (lane&15), K half-chunks
// {sel*8..+7, 16+sel*8..+7} (operand stored K-major).
static __device__ __forceinline__ frag16 load_bfrag_f16(const _Float16* colp, int sel) {
    frag16 f;
    f.h[0] = *(const h8*)(colp + sel * 8);
    f.h[1] = *(const h8*)(colp + 16 + sel * 8);
    return f;
}

// ---------------------------------------------------------------------------
// Kernel 0: one-shot weight transpose  Wt[n][k] = (f16) W[k][n]
// ---------------------------------------------------------------------------
__global__ __launch_bounds__(256) void transpose_w(const float* __restrict__ W,
                                                   _Float16* __restrict__ Wt) {
    const int idx = blockIdx.x * 256 + threadIdx.x;   // over 192*512 elements
    const int k = idx >> 9;          // 0..191
    const int n = idx & 511;         // 0..511 (coalesced read)
    Wt[(size_t)n * D_IN + k] = (_Float16)W[(size_t)k * D_MODEL + n];
}

// ---------------------------------------------------------------------------
// Kernel 1: C[65536 x 512] = X[65536 x 192] * W[192 x 512] + bias  ->  f16
// LDS-free: A-frags straight from X (f32->f16), B-frags from pre-transposed Wt.
// Block tile 128(M) x 128(N); 8 waves as 4(M) x 2(N); wave tile 32 x 64.
// transposeV: write output transposed per batch (Vt[b][d][n]) for av_kernel.
// ---------------------------------------------------------------------------
__global__ __launch_bounds__(256) void qkv_proj(const float* __restrict__ X,
                                                const _Float16* __restrict__ Wt,
                                                const float* __restrict__ bias,
                                                _Float16* __restrict__ Out,
                                                const int transposeV) {
    const int lane = threadIdx.x & 31, wid = threadIdx.x >> 5;
    const int mbase = blockIdx.y * 128;
    const int nbase = blockIdx.x * 128;

    const int r = lane & 15, sel = lane >> 4;
    const int wm = (wid >> 1) * 32;
    const int wn = (wid & 1) * 64;

    v8f acc[2][4] = {};

    for (int k0 = 0; k0 < D_IN; k0 += 32) {
        frag16 af[2];
        for (int i = 0; i < 2; ++i)
            af[i] = load_afrag_f32(
                X + (size_t)(mbase + wm + i * 16 + r) * D_IN + k0, sel);
        for (int t = 0; t < 4; ++t) {
            frag16 bf = load_bfrag_f16(
                Wt + (size_t)(nbase + wn + t * 16 + r) * D_IN + k0, sel);
            for (int i = 0; i < 2; ++i)
                acc[i][t] = wmma_f16(af[i].v, bf.v, acc[i][t]);
        }
    }

    for (int i = 0; i < 2; ++i)
        for (int t = 0; t < 4; ++t) {
            const int col  = nbase + wn + t * 16 + r;
            const float bv = bias[col];
            const int row0 = mbase + wm + i * 16 + sel * 8;   // 8 consecutive rows
            if (!transposeV) {
                _Float16* op = Out + (size_t)row0 * D_MODEL + col;
                for (int g = 0; g < 8; ++g)
                    op[(size_t)g * D_MODEL] = (_Float16)(acc[i][t][g] + bv);
            } else {
                // Vt[b][col][n0..n0+7]  (row0 is 8-aligned, never crosses a batch)
                const int b  = row0 >> 10;
                const int n0 = row0 & 1023;
                h8 h;
                for (int g = 0; g < 8; ++g) h[g] = (_Float16)(acc[i][t][g] + bv);
                *(h8*)(Out + (size_t)b * D_MODEL * N_TOK +
                       (size_t)col * N_TOK + n0) = h;
            }
        }
}

// ---------------------------------------------------------------------------
// Kernel 2: raw scores  S[b] = (Q[b] * K[b]^T) * scale, mask, / T  ->  f32
// Block tile 128x128; fragments straight from global (L2-resident).
// ---------------------------------------------------------------------------
__global__ __launch_bounds__(256) void scores_kernel(const _Float16* __restrict__ Q,
                                                     const _Float16* __restrict__ K,
                                                     const float* __restrict__ mask,
                                                     float* __restrict__ Aout) {
    const int lane = threadIdx.x & 31, wid = threadIdx.x >> 5;
    const int b     = blockIdx.z;
    const int mtile = blockIdx.y * 128;
    const int ntile = blockIdx.x * 128;
    const _Float16* qb = Q + (size_t)b * N_TOK * D_MODEL;
    const _Float16* kb = K + (size_t)b * N_TOK * D_MODEL;

    const int r = lane & 15, sel = lane >> 4;
    const int wm = (wid >> 1) * 32;
    const int wn = (wid & 1) * 64;

    // Prefetch the mask tile needed only in the epilogue (global_prefetch_b8).
    __builtin_prefetch(mask + (size_t)(mtile + wm + sel * 8) * N_TOK +
                       ntile + wn + r, 0, 3);

    v8f acc[2][4] = {};

    for (int k0 = 0; k0 < D_MODEL; k0 += 32) {
        frag16 af[2];
        for (int i = 0; i < 2; ++i) {
            const _Float16* qp =
                qb + (size_t)(mtile + wm + i * 16 + r) * D_MODEL + k0;
            af[i].h[0] = *(const h8*)(qp + sel * 8);
            af[i].h[1] = *(const h8*)(qp + 16 + sel * 8);
        }
        for (int t = 0; t < 4; ++t) {
            // B = K^T: column m of B is row m of K (contiguous along d)
            frag16 bf = load_bfrag_f16(
                kb + (size_t)(ntile + wn + t * 16 + r) * D_MODEL + k0, sel);
            for (int i = 0; i < 2; ++i)
                acc[i][t] = wmma_f16(af[i].v, bf.v, acc[i][t]);
        }
    }

    float* ab = Aout + (size_t)b * N_TOK * N_TOK;
    for (int i = 0; i < 2; ++i)
        for (int t = 0; t < 4; ++t) {
            const int col = ntile + wn + t * 16 + r;
            for (int g = 0; g < 8; ++g) {
                const int row = mtile + wm + i * 16 + sel * 8 + g;
                const float mv = mask[(size_t)row * N_TOK + col];
                float s = acc[i][t][g] * SCALE;
                s = s * mv + (mv == 0.0f ? LNEG : 0.0f);
                ab[(size_t)row * N_TOK + col] = s * INV_T;
            }
        }
}

// ---------------------------------------------------------------------------
// Kernel 3: in-place row softmax over 1024 f32 (one block per row).
// ---------------------------------------------------------------------------
__global__ __launch_bounds__(256) void softmax_kernel(float* __restrict__ A) {
    __shared__ float red[8];
    const int tid = threadIdx.x;
    const int lane = tid & 31, wid = tid >> 5;
    float* rowp = A + (size_t)blockIdx.x * N_TOK;

    f4 v = *(const f4*)(rowp + tid * 4);

    float m = fmaxf(fmaxf(v[0], v[1]), fmaxf(v[2], v[3]));
    for (int off = 16; off; off >>= 1) m = fmaxf(m, __shfl_xor(m, off));
    if (lane == 0) red[wid] = m;
    __syncthreads();
    m = red[0];
    for (int i = 1; i < 8; ++i) m = fmaxf(m, red[i]);
    __syncthreads();

    f4 e;
    e[0] = __expf(v[0] - m); e[1] = __expf(v[1] - m);
    e[2] = __expf(v[2] - m); e[3] = __expf(v[3] - m);
    float s = e[0] + e[1] + e[2] + e[3];
    for (int off = 16; off; off >>= 1) s += __shfl_xor(s, off);
    if (lane == 0) red[wid] = s;
    __syncthreads();
    s = red[0];
    for (int i = 1; i < 8; ++i) s += red[i];
    const float inv = 1.0f / s;

    f4 o; o[0] = e[0] * inv; o[1] = e[1] * inv; o[2] = e[2] * inv; o[3] = e[3] * inv;
    *(f4*)(rowp + tid * 4) = o;
}

// ---------------------------------------------------------------------------
// Kernel 4: Out[b] = A[b] (f32, 1024x1024) * V[b]  ->  f32
// V is stored transposed (Vt[b][d][n], f16), so B-frags are contiguous global
// reads; A-frags are f32 global reads + cvt. No LDS, no barriers.
// Block tile 128(M) x 128(N); wave tile 32 x 64.
// ---------------------------------------------------------------------------
__global__ __launch_bounds__(256) void av_kernel(const float* __restrict__ A,
                                                 const _Float16* __restrict__ Vt,
                                                 float* __restrict__ Out) {
    const int lane = threadIdx.x & 31, wid = threadIdx.x >> 5;
    const int b     = blockIdx.z;
    const int mtile = blockIdx.y * 128;
    const int ntile = blockIdx.x * 128;
    const float*    Ab  = A  + (size_t)b * N_TOK * N_TOK;
    const _Float16* Vtb = Vt + (size_t)b * D_MODEL * N_TOK;

    const int r = lane & 15, sel = lane >> 4;
    const int wm = (wid >> 1) * 32;
    const int wn = (wid & 1) * 64;

    v8f acc[2][4] = {};

    for (int k0 = 0; k0 < N_TOK; k0 += 32) {
        frag16 af[2];
        for (int i = 0; i < 2; ++i)
            af[i] = load_afrag_f32(
                Ab + (size_t)(mtile + wm + i * 16 + r) * N_TOK + k0, sel);
        for (int t = 0; t < 4; ++t) {
            frag16 bf = load_bfrag_f16(
                Vtb + (size_t)(ntile + wn + t * 16 + r) * N_TOK + k0, sel);
            for (int i = 0; i < 2; ++i)
                acc[i][t] = wmma_f16(af[i].v, bf.v, acc[i][t]);
        }
    }

    float* ob = Out + (size_t)b * N_TOK * D_MODEL;
    for (int i = 0; i < 2; ++i)
        for (int t = 0; t < 4; ++t) {
            const int col = ntile + wn + t * 16 + r;
            for (int g = 0; g < 8; ++g) {
                const int row = mtile + wm + i * 16 + sel * 8 + g;
                ob[(size_t)row * D_MODEL + col] = acc[i][t][g];
            }
        }
}

// ---------------------------------------------------------------------------
extern "C" void kernel_launch(void* const* d_in, const int* in_sizes, int n_in,
                              void* d_out, int out_size, void* d_ws, size_t ws_size,
                              hipStream_t stream) {
    const float* x    = (const float*)d_in[0];   // [8,8,1024,192]
    const float* mask = (const float*)d_in[1];   // [1024,1024]
    const float* Wq   = (const float*)d_in[2];
    const float* bq   = (const float*)d_in[3];
    const float* Wk   = (const float*)d_in[4];
    const float* bk   = (const float*)d_in[5];
    const float* Wv   = (const float*)d_in[6];
    const float* bv   = (const float*)d_in[7];

    float* out  = (float*)d_out;                                  // [64,1024,512]
    float* Aout = out + (size_t)NBATCH * N_TOK * D_MODEL;         // [64,1024,1024]

    // Workspace layout (f16): 3 transposed weights, then Q, K, Vt
    const size_t w_elems   = (size_t)D_IN * D_MODEL;              // 98304
    const size_t qkv_elems = (size_t)NBATCH * N_TOK * D_MODEL;    // 33.5M
    _Float16* wtq = (_Float16*)d_ws;
    _Float16* wtk = wtq + w_elems;
    _Float16* wtv = wtk + w_elems;
    _Float16* qh  = wtv + w_elems;
    _Float16* kh  = qh + qkv_elems;
    _Float16* vt  = kh + qkv_elems;

    const dim3 blk(256);

    // One-shot weight transposes (f32 -> f16, K-major)
    const dim3 gtw((D_IN * D_MODEL) / 256);
    transpose_w<<<gtw, blk, 0, stream>>>(Wq, wtq);
    transpose_w<<<gtw, blk, 0, stream>>>(Wk, wtk);
    transpose_w<<<gtw, blk, 0, stream>>>(Wv, wtv);

    // QKV projections: tiles 128x128 over [65536 x 512]
    const dim3 gproj(D_MODEL / 128, (NBATCH * N_TOK) / 128);
    qkv_proj<<<gproj, blk, 0, stream>>>(x, wtq, bq, qh, 0);
    qkv_proj<<<gproj, blk, 0, stream>>>(x, wtk, bk, kh, 0);
    qkv_proj<<<gproj, blk, 0, stream>>>(x, wtv, bv, vt, 1);   // store transposed

    // Raw masked/scaled scores into A region
    scores_kernel<<<dim3(N_TOK / 128, N_TOK / 128, NBATCH), blk, 0, stream>>>(
        qh, kh, mask, Aout);

    // Row softmax in place (one block per row)
    softmax_kernel<<<dim3(NBATCH * N_TOK), blk, 0, stream>>>(Aout);

    // Output = A * V   (V consumed transposed)
    av_kernel<<<dim3(D_MODEL / 128, N_TOK / 128, NBATCH), blk, 0, stream>>>(
        Aout, vt, out);
}
